// VALSTM_27530740367563
// MI455X (gfx1250) — matrix operand order repaired and checked by reading, compiled, not run on hardware
//
#include <hip/hip_runtime.h>
#include <hip/hip_bf16.h>

// ---------------------------------------------------------------------------
// VALSTM fused pipeline for gfx1250 (MI455X): all GEMMs on v_wmma_f32_16x16x32_bf16
// Round 2: ping-pong h panels (no accumulators live across barriers -> no spills),
// vector 16B A-fragment loads (no per-element guards), stride-128 seq buffers,
// v_tanh_f32 fast path.
// ---------------------------------------------------------------------------

typedef __attribute__((ext_vector_type(16))) __bf16 v16bf;
typedef __attribute__((ext_vector_type(8)))  __bf16 v8bf;
typedef __attribute__((ext_vector_type(8)))  float  v8f;

#define B_  1024
#define T_  128
#define H_  100
#define NF_ 150
#define NCLS_ 60

// K layout: x region padded to 160 (k-tiles 0..4), h region padded to 128
// (k-tiles 5..8). N layout: gates padded 100->112, N=448, gate g at n-tiles [7g,7g+7).
#define KT_  9
#define NT_  28
#define FRAG_ 512   // halves per packed B fragment (32 lanes x 16)

constexpr long BT_   = (long)B_ * T_;
constexpr long OFF_XBF   = 0;                          // BT*160 bf16
constexpr long OFF_X3D   = OFF_XBF  + BT_*160*2;       // BT*150 f32
constexpr long OFF_XTR   = OFF_X3D  + BT_*150*4;       // BT*160 bf16
constexpr long OFF_SEQA  = OFF_XTR  + BT_*160*2;       // BT*128 bf16
constexpr long OFF_SEQB  = OFF_SEQA + BT_*128*2;       // BT*128 bf16
constexpr long OFF_WLSTM = OFF_SEQB + BT_*128*2;       // 5 sets * KT*NT*FRAG bf16
constexpr long WSET_ELEMS = (long)KT_*NT_*FRAG_;       // 129024
constexpr long OFF_WFC   = OFF_WLSTM + 5*WSET_ELEMS*2; // 512*4*512 bf16
constexpr long OFF_LOGITS= OFF_WFC  + 512L*4*FRAG_*2;  // 1024*64 f32

#if __has_builtin(__builtin_amdgcn_tanhf)
__device__ inline float tanh_fast(float x) { return __builtin_amdgcn_tanhf(x); }
#else
__device__ inline float tanh_fast(float x) { return 1.0f - 2.0f / (__expf(2.0f * x) + 1.0f); }
#endif
__device__ inline float sigf(float x) { return 1.0f / (1.0f + __expf(-x)); }

__device__ inline v8f wmma_bf16(v16bf a, v16bf b, v8f c) {
  return __builtin_amdgcn_wmma_f32_16x16x32_bf16(false, a, false, b, (short)0, c,
                                                 false, false);
}

// A fragment: lanerow = row pointer for this lane (already includes (lane&15)*rstride).
// Lane-group g needs K = kbase+g*16 .. +15 (contiguous, matching packed B):
// two contiguous 16-byte loads + concat. Requires rows 16B-aligned (all strides mult. of 8).
__device__ inline v16bf ldA2(const __bf16* lanerow, int kbase) {
  int g = (threadIdx.x >> 4) & 1;
  const __bf16* p = lanerow + kbase + g * 16;
  v8bf lo = *(const v8bf*)p;
  v8bf hi = *(const v8bf*)(p + 8);
  return __builtin_shufflevector(lo, hi, 0, 1, 2, 3, 4, 5, 6, 7, 8, 9, 10, 11, 12, 13, 14, 15);
}

// B fragment pre-packed frag-major: 32 bytes per lane, contiguous.
__device__ inline v16bf ldB(const __bf16* frag) {
  int lane = threadIdx.x & 31;
  return *(const v16bf*)(frag + lane * 16);
}

// ---------------------------------------------------------------------------
// Kernel 1: transpose X_skeleton (B,3,T,25,2) -> Xbf bf16 (B,T,160) + X3D f32 (B,T,150)
// ---------------------------------------------------------------------------
__global__ void k_prep(const float* __restrict__ Xs, __bf16* __restrict__ Xbf,
                       float* __restrict__ X3D) {
  long gid = (long)blockIdx.x * blockDim.x + threadIdx.x;
  if (gid >= BT_ * 160) return;
  int f = (int)(gid % 160);
  long bt = gid / 160;
  int t = (int)(bt % T_);
  long b = bt / T_;
  if (f < NF_) {
    int c = f % 3, jd = f / 3, j = jd >> 1, d = jd & 1;
    float v = Xs[(((b * 3 + c) * T_ + t) * 25 + j) * 2 + d];
    Xbf[gid] = (__bf16)v;
    X3D[bt * NF_ + f] = v;
  } else {
    Xbf[gid] = (__bf16)0.0f;
  }
}

// ---------------------------------------------------------------------------
// Kernel 2: pack one LSTM weight set [Wih | Whh] into WMMA B-fragment order.
// K<160 -> x (valid K<in_f); K in [160,260) -> h. gate = N/112, idx = N%112.
// ---------------------------------------------------------------------------
__global__ void k_packw(const float* __restrict__ Wih, const float* __restrict__ Whh,
                        int in_f, __bf16* __restrict__ dst) {
  int gid = blockIdx.x * blockDim.x + threadIdx.x;
  if (gid >= KT_ * NT_ * FRAG_) return;
  int j = gid & 15;
  int lane = (gid >> 4) & 31;
  int nt = (gid >> 9) % NT_;
  int kt = gid / (NT_ * FRAG_);
  int N = nt * 16 + (lane & 15);
  int K = kt * 32 + (lane >> 4) * 16 + j;
  int gate = N / 112, idx = N % 112;
  float v = 0.0f;
  if (idx < H_) {
    int nsrc = gate * H_ + idx;
    if (K < 160) {
      if (K < in_f) v = Wih[nsrc * in_f + K];
    } else {
      int kh = K - 160;
      if (kh < H_) v = Whh[nsrc * H_ + kh];
    }
  }
  dst[gid] = (__bf16)v;
}

// ---------------------------------------------------------------------------
// Kernel 3: pack W_fc (60 x 12800) -> fragments over K' = t*128 + h (K'=16384)
// ---------------------------------------------------------------------------
__global__ void k_packwfc(const float* __restrict__ Wfc, __bf16* __restrict__ dst) {
  int gid = blockIdx.x * blockDim.x + threadIdx.x;
  if (gid >= 512 * 4 * FRAG_) return;
  int j = gid & 15;
  int lane = (gid >> 4) & 31;
  int nt = (gid >> 9) & 3;
  int kt = gid / (4 * FRAG_);
  int N = nt * 16 + (lane & 15);
  int K = kt * 32 + (lane >> 4) * 16 + j;
  int t = K >> 7, h = K & 127;
  float v = 0.0f;
  if (N < NCLS_ && h < H_) v = Wfc[N * (H_ * T_) + t * H_ + h];
  dst[gid] = (__bf16)v;
}

// ---------------------------------------------------------------------------
// Per-row FC rot/tr + R@R@R rotation + joint transform for one (b,t).
// ---------------------------------------------------------------------------
__device__ inline void rot_row(long b, int t, const __bf16* hr, const __bf16* ht,
                               const float* __restrict__ Wfcr, const float* __restrict__ bfcr,
                               const float* __restrict__ Wfct, const float* __restrict__ bfct,
                               const float* __restrict__ X3D, __bf16* __restrict__ Xtr) {
  float rp[3], tr[3];
#pragma unroll
  for (int c = 0; c < 3; ++c) {
    float ar = bfcr[c], at = bfct[c];
    for (int k = 0; k < H_; ++k) {
      ar += Wfcr[c * H_ + k] * (float)hr[k];
      at += Wfct[c * H_ + k] * (float)ht[k];
    }
    rp[c] = ar; tr[c] = at;
  }
  float sx = __sinf(rp[0]);
  float sy = __sinf(rp[1]), cy = __cosf(rp[1]);
  float sz = __sinf(rp[2]), cz = __cosf(rp[2]);
  float R[9] = {cz, -sz, sy, sz, cz, -sx, -sy, sx, cy};
  float R2[9], R3[9];
#pragma unroll
  for (int i = 0; i < 3; ++i)
#pragma unroll
    for (int jc = 0; jc < 3; ++jc)
      R2[i * 3 + jc] = R[i * 3 + 0] * R[0 * 3 + jc] + R[i * 3 + 1] * R[1 * 3 + jc] +
                       R[i * 3 + 2] * R[2 * 3 + jc];
#pragma unroll
  for (int i = 0; i < 3; ++i)
#pragma unroll
    for (int jc = 0; jc < 3; ++jc)
      R3[i * 3 + jc] = R2[i * 3 + 0] * R[0 * 3 + jc] + R2[i * 3 + 1] * R[1 * 3 + jc] +
                       R2[i * 3 + 2] * R[2 * 3 + jc];
  const float* xin = X3D + (b * T_ + t) * NF_;
  __bf16* xo = Xtr + (b * T_ + t) * 160;
  for (int j = 0; j < 50; ++j) {
    float v0 = xin[3 * j + 0] - tr[0];
    float v1 = xin[3 * j + 1] - tr[1];
    float v2 = xin[3 * j + 2] - tr[2];
#pragma unroll
    for (int i = 0; i < 3; ++i)
      xo[3 * j + i] = (__bf16)(R3[i * 3 + 0] * v0 + R3[i * 3 + 1] * v1 + R3[i * 3 + 2] * v2);
  }
#pragma unroll
  for (int p = NF_; p < 160; ++p) xo[p] = (__bf16)0.0f;
}

// ---------------------------------------------------------------------------
// Kernel 4: Phase-1 scan. 16 WGs x 896 threads (28 waves, 2 units each).
// Units: 2 cells x 4 m-tiles x 7 h-tiles = 56. c state in VGPRs for the whole
// scan; h ping-pongs between two LDS panels (one barrier per step, accumulators
// never live across a barrier). Rotation for step t-1 overlaps step t's GEMM.
// ---------------------------------------------------------------------------
__global__ __launch_bounds__(896) void k_scan1(
    const __bf16* __restrict__ Xbf, const float* __restrict__ X3D,
    const __bf16* __restrict__ Wf_rot, const __bf16* __restrict__ Wf_tr,
    const float* __restrict__ bih_rot, const float* __restrict__ bhh_rot,
    const float* __restrict__ bih_tr, const float* __restrict__ bhh_tr,
    const float* __restrict__ Wfcr, const float* __restrict__ bfcr,
    const float* __restrict__ Wfct, const float* __restrict__ bfct,
    __bf16* __restrict__ Xtr) {
  __shared__ __bf16 hpan[2][2][64][128];  // [buf][cell][row][col], 64 KB
  const int tid = threadIdx.x;
  const int lane = tid & 31;
  const int wave = tid >> 5;
  const int b0 = blockIdx.x * 64;

  for (int i = tid; i < 2 * 2 * 64 * 128; i += 896) ((__bf16*)hpan)[i] = (__bf16)0.0f;

  int cell_[2], mt_[2], hjt_[2];
  float bias[2][4];
  float creg[2][8];
#pragma unroll
  for (int s = 0; s < 2; ++s) {
    int u = wave * 2 + s;
    int cell = u / 28, rem = u % 28;
    cell_[s] = cell; mt_[s] = rem / 7; hjt_[s] = rem % 7;
    int idx = hjt_[s] * 16 + (lane & 15);
    const float* bi = cell ? bih_tr : bih_rot;
    const float* bh = cell ? bhh_tr : bhh_rot;
#pragma unroll
    for (int g = 0; g < 4; ++g)
      bias[s][g] = (idx < H_) ? (bi[g * H_ + idx] + bh[g * H_ + idx]) : 0.0f;
#pragma unroll
    for (int r = 0; r < 8; ++r) creg[s][r] = 0.0f;
  }
  __syncthreads();

  for (int t = 0; t < T_; ++t) {
    const int rb = t & 1, wb = rb ^ 1;
    // rotation for the PREVIOUS step overlaps this step's GEMM (reads rb only)
    if (t > 0 && tid < 64)
      rot_row(b0 + tid, t - 1, &hpan[rb][0][tid][0], &hpan[rb][1][tid][0],
              Wfcr, bfcr, Wfct, bfct, X3D, Xtr);
#pragma unroll
    for (int s = 0; s < 2; ++s) {
      const int cell = cell_[s], mt = mt_[s], hjt = hjt_[s];
      const __bf16* wset = cell ? Wf_tr : Wf_rot;
      const __bf16* wh = wset + hjt * FRAG_;
      const __bf16* xrow = Xbf + ((long)(b0 + mt * 16 + (lane & 15)) * T_ + t) * 160;
      const __bf16* hrow = &hpan[rb][cell][mt * 16 + (lane & 15)][0];
      v8f ai = {}, af = {}, ag = {}, ao = {};
#pragma unroll
      for (int kk = 0; kk < 5; ++kk) {
        v16bf a = ldA2(xrow, kk * 32);
        const __bf16* wk = wh + kk * NT_ * FRAG_;
        ai = wmma_bf16(a, ldB(wk + 0 * 7 * FRAG_), ai);
        af = wmma_bf16(a, ldB(wk + 1 * 7 * FRAG_), af);
        ag = wmma_bf16(a, ldB(wk + 2 * 7 * FRAG_), ag);
        ao = wmma_bf16(a, ldB(wk + 3 * 7 * FRAG_), ao);
      }
#pragma unroll
      for (int kh = 0; kh < 4; ++kh) {
        v16bf a = ldA2(hrow, kh * 32);
        const __bf16* wk = wh + (5 + kh) * NT_ * FRAG_;
        ai = wmma_bf16(a, ldB(wk + 0 * 7 * FRAG_), ai);
        af = wmma_bf16(a, ldB(wk + 1 * 7 * FRAG_), af);
        ag = wmma_bf16(a, ldB(wk + 2 * 7 * FRAG_), ag);
        ao = wmma_bf16(a, ldB(wk + 3 * 7 * FRAG_), ao);
      }
      // gate nonlinearities immediately (write goes to the OTHER buffer)
      const int n = hjt * 16 + (lane & 15);
      const int gsel = (lane >> 4) & 1;
#pragma unroll
      for (int r = 0; r < 8; ++r) {
        float ig = sigf(ai[r] + bias[s][0]);
        float fg = sigf(af[r] + bias[s][1]);
        float gg = tanh_fast(ag[r] + bias[s][2]);
        float og = sigf(ao[r] + bias[s][3]);
        float cc = fg * creg[s][r] + ig * gg;
        creg[s][r] = cc;
        float hv = og * tanh_fast(cc);
        hpan[wb][cell][mt * 16 + r + 8 * gsel][n] = (n < H_) ? (__bf16)hv : (__bf16)0.0f;
      }
    }
    __syncthreads();
  }
  // final rotation for t = 127 (h^127 lives in buffer 0)
  if (tid < 64)
    rot_row(b0 + tid, T_ - 1, &hpan[0][0][tid][0], &hpan[0][1][tid][0],
            Wfcr, bfcr, Wfct, bfct, X3D, Xtr);
}

// ---------------------------------------------------------------------------
// Kernel 5: Phase-2 scan (one layer), serial over the BATCH axis (faithful to
// the reference's nn.LSTM misuse). One persistent WG, 28 waves x 2 units
// (8 m-tiles x 7 h-tiles). KX = number of x k-tiles (5 for in_f=150, 4 for 100).
// src stride SW is 160 (layer 0) or 128; dst stride is 128.
// ---------------------------------------------------------------------------
template <int KX, int SW>
__global__ __launch_bounds__(896) void k_scan2(
    const __bf16* __restrict__ src, const __bf16* __restrict__ Wf,
    const float* __restrict__ bih, const float* __restrict__ bhh,
    __bf16* __restrict__ dst) {
  __shared__ __bf16 hpan[2][128][128];  // 64 KB
  const int tid = threadIdx.x;
  const int lane = tid & 31;
  const int wave = tid >> 5;

  for (int i = tid; i < 2 * 128 * 128; i += 896) ((__bf16*)hpan)[i] = (__bf16)0.0f;

  int mt_[2], hjt_[2];
  float bias[2][4];
  float creg[2][8];
#pragma unroll
  for (int s = 0; s < 2; ++s) {
    int u = wave * 2 + s;
    mt_[s] = u / 7; hjt_[s] = u % 7;
    int idx = hjt_[s] * 16 + (lane & 15);
#pragma unroll
    for (int g = 0; g < 4; ++g)
      bias[s][g] = (idx < H_) ? (bih[g * H_ + idx] + bhh[g * H_ + idx]) : 0.0f;
#pragma unroll
    for (int r = 0; r < 8; ++r) creg[s][r] = 0.0f;
  }
  __syncthreads();

  for (int b = 0; b < B_; ++b) {
    const int rb = b & 1, wb = rb ^ 1;
#pragma unroll
    for (int s = 0; s < 2; ++s) {
      const int mt = mt_[s], hjt = hjt_[s];
      const __bf16* wh = Wf + hjt * FRAG_;
      const __bf16* xrow = src + ((long)b * T_ + mt * 16 + (lane & 15)) * SW;
      const __bf16* hrow = &hpan[rb][mt * 16 + (lane & 15)][0];
      v8f ai = {}, af = {}, ag = {}, ao = {};
#pragma unroll
      for (int kk = 0; kk < KX; ++kk) {
        v16bf a = ldA2(xrow, kk * 32);
        const __bf16* wk = wh + kk * NT_ * FRAG_;
        ai = wmma_bf16(a, ldB(wk + 0 * 7 * FRAG_), ai);
        af = wmma_bf16(a, ldB(wk + 1 * 7 * FRAG_), af);
        ag = wmma_bf16(a, ldB(wk + 2 * 7 * FRAG_), ag);
        ao = wmma_bf16(a, ldB(wk + 3 * 7 * FRAG_), ao);
      }
#pragma unroll
      for (int kh = 0; kh < 4; ++kh) {
        v16bf a = ldA2(hrow, kh * 32);
        const __bf16* wk = wh + (5 + kh) * NT_ * FRAG_;
        ai = wmma_bf16(a, ldB(wk + 0 * 7 * FRAG_), ai);
        af = wmma_bf16(a, ldB(wk + 1 * 7 * FRAG_), af);
        ag = wmma_bf16(a, ldB(wk + 2 * 7 * FRAG_), ag);
        ao = wmma_bf16(a, ldB(wk + 3 * 7 * FRAG_), ao);
      }
      const int n = hjt * 16 + (lane & 15);
      const int gsel = (lane >> 4) & 1;
#pragma unroll
      for (int r = 0; r < 8; ++r) {
        float ig = sigf(ai[r] + bias[s][0]);
        float fg = sigf(af[r] + bias[s][1]);
        float gg = tanh_fast(ag[r] + bias[s][2]);
        float og = sigf(ao[r] + bias[s][3]);
        float cc = fg * creg[s][r] + ig * gg;
        creg[s][r] = cc;
        float hv = og * tanh_fast(cc);
        __bf16 hb16 = (n < H_) ? (__bf16)hv : (__bf16)0.0f;
        int trow = mt * 16 + r + 8 * gsel;
        hpan[wb][trow][n] = hb16;
        dst[((long)b * T_ + trow) * 128 + n] = hb16;
      }
    }
    __syncthreads();
  }
}

// ---------------------------------------------------------------------------
// Kernel 6: final FC on WMMA. 64 WGs (one 16-row m-tile) x 4 waves (one 16-col
// n-tile). K = 512 k-tiles over (t*128+h); zero weights kill pad columns.
// ---------------------------------------------------------------------------
__global__ __launch_bounds__(128) void k_fc(const __bf16* __restrict__ seq,
                                            const __bf16* __restrict__ wf,
                                            const float* __restrict__ bfc,
                                            float* __restrict__ logits) {
  const int lane = threadIdx.x & 31;
  const int wave = threadIdx.x >> 5;
  const int mt = blockIdx.x;
  const __bf16* lrow = seq + ((long)mt * 16 + (lane & 15)) * (T_ * 128);
  v8f acc = {};
  for (int kt = 0; kt < 512; ++kt) {
    v16bf a = ldA2(lrow, kt * 32);
    acc = wmma_bf16(a, ldB(wf + (kt * 4 + wave) * FRAG_), acc);
  }
  const int col = wave * 16 + (lane & 15);
  const int gsel = (lane >> 4) & 1;
  const float bc = (col < NCLS_) ? bfc[col] : 0.0f;
#pragma unroll
  for (int r = 0; r < 8; ++r)
    logits[(mt * 16 + r + 8 * gsel) * 64 + col] = acc[r] + bc;
}

// ---------------------------------------------------------------------------
// Kernel 7: log-softmax over 60 classes, one thread per row.
// ---------------------------------------------------------------------------
__global__ void k_lsm(const float* __restrict__ logits, float* __restrict__ out) {
  int row = blockIdx.x * blockDim.x + threadIdx.x;
  if (row >= B_) return;
  const float* L = logits + row * 64;
  float m = -1e30f;
  for (int c = 0; c < NCLS_; ++c) m = fmaxf(m, L[c]);
  float sum = 0.0f;
  for (int c = 0; c < NCLS_; ++c) sum += __expf(L[c] - m);
  float ls = __logf(sum);
  for (int c = 0; c < NCLS_; ++c) out[row * NCLS_ + c] = L[c] - m - ls;
}

// ---------------------------------------------------------------------------
extern "C" void kernel_launch(void* const* d_in, const int* in_sizes, int n_in,
                              void* d_out, int out_size, void* d_ws, size_t ws_size,
                              hipStream_t stream) {
  const float* Xs      = (const float*)d_in[0];
  const float* Wih_rot = (const float*)d_in[2];
  const float* Whh_rot = (const float*)d_in[3];
  const float* bih_rot = (const float*)d_in[4];
  const float* bhh_rot = (const float*)d_in[5];
  const float* Wih_tr  = (const float*)d_in[6];
  const float* Whh_tr  = (const float*)d_in[7];
  const float* bih_tr  = (const float*)d_in[8];
  const float* bhh_tr  = (const float*)d_in[9];
  const float* Wfcr    = (const float*)d_in[10];
  const float* bfcr    = (const float*)d_in[11];
  const float* Wfct    = (const float*)d_in[12];
  const float* bfct    = (const float*)d_in[13];
  const float* lWih[3] = {(const float*)d_in[14], (const float*)d_in[15], (const float*)d_in[16]};
  const float* lWhh[3] = {(const float*)d_in[17], (const float*)d_in[18], (const float*)d_in[19]};
  const float* lbih[3] = {(const float*)d_in[20], (const float*)d_in[21], (const float*)d_in[22]};
  const float* lbhh[3] = {(const float*)d_in[23], (const float*)d_in[24], (const float*)d_in[25]};
  const float* Wfc     = (const float*)d_in[26];
  const float* bfc     = (const float*)d_in[27];

  char* ws = (char*)d_ws;
  __bf16* Xbf    = (__bf16*)(ws + OFF_XBF);
  float*  X3D    = (float*)(ws + OFF_X3D);
  __bf16* Xtr    = (__bf16*)(ws + OFF_XTR);
  __bf16* SeqA   = (__bf16*)(ws + OFF_SEQA);
  __bf16* SeqB   = (__bf16*)(ws + OFF_SEQB);
  __bf16* Wl     = (__bf16*)(ws + OFF_WLSTM);
  __bf16* WfcF   = (__bf16*)(ws + OFF_WFC);
  float*  logits = (float*)(ws + OFF_LOGITS);

  {
    long n = BT_ * 160;
    k_prep<<<(unsigned)((n + 255) / 256), 256, 0, stream>>>(Xs, Xbf, X3D);
  }
  {
    int n = KT_ * NT_ * FRAG_;
    unsigned g = (n + 255) / 256;
    k_packw<<<g, 256, 0, stream>>>(Wih_rot, Whh_rot, NF_, Wl + 0 * WSET_ELEMS);
    k_packw<<<g, 256, 0, stream>>>(Wih_tr,  Whh_tr,  NF_, Wl + 1 * WSET_ELEMS);
    k_packw<<<g, 256, 0, stream>>>(lWih[0], lWhh[0], NF_, Wl + 2 * WSET_ELEMS);
    k_packw<<<g, 256, 0, stream>>>(lWih[1], lWhh[1], H_,  Wl + 3 * WSET_ELEMS);
    k_packw<<<g, 256, 0, stream>>>(lWih[2], lWhh[2], H_,  Wl + 4 * WSET_ELEMS);
  }
  {
    int n = 512 * 4 * FRAG_;
    k_packwfc<<<(unsigned)((n + 255) / 256), 256, 0, stream>>>(Wfc, WfcF);
  }
  k_scan1<<<16, 896, 0, stream>>>(Xbf, X3D, Wl + 0 * WSET_ELEMS, Wl + 1 * WSET_ELEMS,
                                  bih_rot, bhh_rot, bih_tr, bhh_tr,
                                  Wfcr, bfcr, Wfct, bfct, Xtr);
  k_scan2<5, 160><<<1, 896, 0, stream>>>(Xtr,  Wl + 2 * WSET_ELEMS, lbih[0], lbhh[0], SeqA);
  k_scan2<4, 128><<<1, 896, 0, stream>>>(SeqA, Wl + 3 * WSET_ELEMS, lbih[1], lbhh[1], SeqB);
  k_scan2<4, 128><<<1, 896, 0, stream>>>(SeqB, Wl + 4 * WSET_ELEMS, lbih[2], lbhh[2], SeqA);
  k_fc<<<64, 128, 0, stream>>>(SeqA, WfcF, bfc, logits);
  k_lsm<<<4, 256, 0, stream>>>(logits, (float*)d_out);
}
